// HarmonicModel_38611755991678
// MI455X (gfx1250) — compile-verified, matrix operand order checked by reading, not run on hardware
//
#include <hip/hip_runtime.h>
#include <math.h>

#define N_AUDIO   8
#define N_HARM    100
#define N_FRAMES  1000
#define N_SAMPLES 64000
#define TWO_PI_F     6.28318530717958647692f
#define INV_TWO_PI_F 0.15915494309189533577f
#define PI_F         3.14159265358979323846f
#define OMEGA_K      (TWO_PI_F / 16000.0f)   // 2*pi / SAMPLE_RATE

typedef __attribute__((ext_vector_type(16))) _Float16 v16h;
typedef __attribute__((ext_vector_type(8)))  float    v8f;

__device__ __forceinline__ float wrap_pm_half(float d) {
    d = (d >  0.5f) ? d - 1.0f : d;
    d = (d < -0.5f) ? d + 1.0f : d;
    return d;
}

__device__ __forceinline__ float wrap_2pi(float x) {
    return x - TWO_PI_F * floorf(x * INV_TWO_PI_F);
}

// ---------------------------------------------------------------------------
// Kernel A: per (audio, harmonic) series, sequential over 1000 frames.
// Computes per-frame omega_f and the per-frame starting phase
// prefix_f = (sum of phase increments of frames 0..f-1) mod 2*pi.
// Frame increment (piecewise-linear freq within a 64-sample frame):
//   S_f = sum_{j=0..63} (w_f + (j/64)*(w_{f+1}-w_f)) = 64*w_f + 31.5*(w_{f+1}-w_f)
// 800 threads total, each ~1000 cheap iterations; loads are h-contiguous
// (32 consecutive harmonics per wave) so fully coalesced.
// ---------------------------------------------------------------------------
__global__ void hm_scan_kernel(const float* __restrict__ hfreq,   // (8,1000,100)
                               const float* __restrict__ ps,      // (8,1001,100)
                               float* __restrict__ omega,         // (8,1000,100)
                               float* __restrict__ prefix)        // (8,1000,100)
{
    int h = threadIdx.x;
    int b = blockIdx.x;
    if (h >= N_HARM) return;

    const float* psb = ps    + (size_t)b * (N_FRAMES + 1) * N_HARM + h;
    const float* hfb = hfreq + (size_t)b * N_FRAMES * N_HARM + h;
    float*       omb = omega + (size_t)b * N_FRAMES * N_HARM + h;
    float*       prb = prefix+ (size_t)b * N_FRAMES * N_HARM + h;

    float ps_prev = psb[0];
    float ps_next = psb[N_HARM];
    float d       = wrap_pm_half(ps_next - ps_prev);
    float om_prev = (hfb[0] + 250.0f * d) * OMEGA_K;  // 250 = SAMPLE_RATE/FRAME_STEP
    omb[0] = om_prev;
    prb[0] = 0.0f;
    float running = 0.0f;
    ps_prev = ps_next;

    for (int f = 1; f < N_FRAMES; ++f) {
        ps_next = psb[(size_t)(f + 1) * N_HARM];
        d = wrap_pm_half(ps_next - ps_prev);
        float om = (hfb[(size_t)f * N_HARM] + 250.0f * d) * OMEGA_K;
        float S  = 64.0f * om_prev + 31.5f * (om - om_prev);  // increment of frame f-1
        running  = wrap_2pi(running + S);
        omb[(size_t)f * N_HARM] = om;
        prb[(size_t)f * N_HARM] = running;
        om_prev = om;
        ps_prev = ps_next;
    }
}

// ---------------------------------------------------------------------------
// Kernel B: one 128-thread block per (audio, frame). Each wave32 synthesizes
// 16 samples; the harmonic reduction is done on the matrix pipe:
//   D(16x16,f32) += A(16x32 f16: amp*sin terms) x B(32x16 f16: ones)
// 4 chunks cover 128 padded harmonics. Column N=0 of D = per-sample sums.
// ---------------------------------------------------------------------------
__global__ __launch_bounds__(128)
void hm_synth_kernel(const float* __restrict__ ps,      // (8,1001,100)
                     const float* __restrict__ amps,    // (8,1000,100)
                     const float* __restrict__ omega,   // (8,1000,100)
                     const float* __restrict__ prefix,  // (8,1000,100)
                     float* __restrict__ out)           // (8,64000)
{
    __shared__ float sOm[N_HARM], sOmN[N_HARM], sPre[N_HARM];
    __shared__ float sA0[N_HARM], sA1[N_HARM], sIp[N_HARM];

    const int f   = blockIdx.x;
    const int b   = blockIdx.y;
    const int tid = threadIdx.x;
    const int fn  = (f + 1 < N_FRAMES) ? (f + 1) : (N_FRAMES - 1);

    const size_t base  = ((size_t)b * N_FRAMES + f)  * N_HARM;
    const size_t basen = ((size_t)b * N_FRAMES + fn) * N_HARM;

    if (tid < N_HARM) {
        sOm[tid]  = omega [base  + tid];
        sOmN[tid] = omega [basen + tid];
        sPre[tid] = prefix[base  + tid];
        sA0[tid]  = fabsf(amps[base  + tid]);
        sA1[tid]  = fabsf(amps[basen + tid]);
        sIp[tid]  = ps[(size_t)b * (N_FRAMES + 1) * N_HARM + tid]; // initial phase
    }
    __syncthreads();

    const int lane  = tid & 31;            // wave32
    const int wave  = tid >> 5;            // 4 waves per block
    const int hb    = (lane >> 4) & 1;     // K-half within A layout
    const int msamp = (wave << 4) + (lane & 15);   // sample offset 0..63 in frame

    const float fm   = (float)msamp;
    const float frac = fm * (1.0f / 64.0f);
    const float mp1  = fm + 1.0f;
    const float tri  = fm * mp1 * (1.0f / 128.0f); // m(m+1)/(2*64)

    v16h ones;
    #pragma unroll
    for (int i = 0; i < 16; ++i) ones[i] = (_Float16)1.0f;

    v8f acc = {};

    #pragma unroll
    for (int c = 0; c < 4; ++c) {          // 4 x 32 = 128 padded harmonics
        v16h a;
        #pragma unroll
        for (int v = 0; v < 8; ++v) {
            // ISA 16-bit A 16x32 layout: lanes 0-15 hold K 0-7 (V0-3), 16-23 (V4-7);
            // lanes 16-31 hold K 8-15, 24-31. Two packed f16 per VGPR.
            const int k0 = ((v < 4) ? (2 * v) : (16 + 2 * (v - 4))) + 8 * hb;
            #pragma unroll
            for (int q = 0; q < 2; ++q) {
                const int h  = 32 * c + k0 + q;
                const int hc = (h < N_HARM) ? h : 0;           // branch-free masking
                const float mask = (h < N_HARM) ? 1.0f : 0.0f; // keeps EXEC all-ones
                const float wa = sOm[hc], wb = sOmN[hc];
                const float dw = wb - wa;
                const float wenv = wa + frac * dw;             // omega envelope
                float aenv = sA0[hc] + frac * (sA1[hc] - sA0[hc]);
                aenv = (wenv < PI_F) ? aenv : 0.0f;            // freq_env < Nyquist
                float P = sPre[hc] + mp1 * wa + tri * dw;      // analytic cumsum
                P = wrap_2pi(P) + sIp[hc];
                const float x = mask * aenv * __sinf(P);
                a[2 * v + q] = (_Float16)x;
            }
        }
        acc = __builtin_amdgcn_wmma_f32_16x16x32_f16(
                  false, a, false, ones, (short)0, acc, false, false);
    }

    // D layout: VGPR r, lanes 0-15 -> M=r, lanes 16-31 -> M=r+8; all N columns
    // are identical (B = ones), so lanes with N==0 (lanes 0 and 16) hold the
    // 16 per-sample sums in 8 consecutive VGPRs each.
    if ((lane & 15) == 0) {
        const size_t o = (size_t)b * N_SAMPLES + (size_t)f * 64
                       + (size_t)(wave << 4) + (size_t)(hb << 3);
        #pragma unroll
        for (int r = 0; r < 8; ++r) out[o + r] = acc[r];
    }
}

// ---------------------------------------------------------------------------
extern "C" void kernel_launch(void* const* d_in, const int* in_sizes, int n_in,
                              void* d_out, int out_size, void* d_ws, size_t ws_size,
                              hipStream_t stream) {
    (void)in_sizes; (void)n_in; (void)out_size; (void)ws_size;
    // d_in[0] = audio (unused by the reference)
    const float* hfreq = (const float*)d_in[1];   // (8,1000,100)
    const float* ps    = (const float*)d_in[2];   // (8,1001,100)
    const float* amps  = (const float*)d_in[3];   // (8,1000,100)
    float* out = (float*)d_out;                   // (8,64000)

    float* omega  = (float*)d_ws;                                   // 3.2 MB
    float* prefix = omega + (size_t)N_AUDIO * N_FRAMES * N_HARM;    // 3.2 MB

    hm_scan_kernel<<<dim3(N_AUDIO), dim3(128), 0, stream>>>(hfreq, ps, omega, prefix);
    hm_synth_kernel<<<dim3(N_FRAMES, N_AUDIO), dim3(128), 0, stream>>>(
        ps, amps, omega, prefix, out);
}
